// LoRAAttention_3049426780207
// MI455X (gfx1250) — matrix operand order repaired and checked by reading, not compiled, not run on hardware
//
#include <hip/hip_runtime.h>

typedef __attribute__((ext_vector_type(16))) __bf16 v16bf;
typedef __attribute__((ext_vector_type(8)))  __bf16 v8bf;
typedef __attribute__((ext_vector_type(8)))  float  v8f;

#define NH 12
#define HD 64
#define CC 768
#define HWD 1024
#define BB 8
#define BHD (BB*NH)        // 96
#define MROWS (BB*HWD)     // 8192
#define NQKV (3*CC)        // 2304
#define LORA_SCALE 4.0f
#define ATT_SCALE 0.125f

#define WMMA_BF16(A,B,C) __builtin_amdgcn_wmma_f32_16x16x32_bf16(false,(A),false,(B),(short)0,(C),false,false)

// A-operand fragment (16x32, row per lane): lane<16 -> K {0..7,16..23}, lane>=16 -> K {8..15,24..31}
static __device__ __forceinline__ v16bf load_fragA(const __bf16* base, int ld, int row, int kb, int lane) {
  int hi = (lane >> 4) & 1;
  const __bf16* p = base + (size_t)row * ld + kb + hi * 8;
  v8bf lo = *(const v8bf*)(p);
  v8bf hh = *(const v8bf*)(p + 16);
  return __builtin_shufflevector(lo, hh, 0,1,2,3,4,5,6,7,8,9,10,11,12,13,14,15);
}

// B-operand fragment (32x16, column per lane): lane<16 -> K {0..7,16..23}, lane>=16 -> K {8..15,24..31}
static __device__ __forceinline__ v16bf load_fragB(const __bf16* base, int ld, int col, int kb, int lane) {
  int hi = (lane >> 4) & 1;
  const __bf16* p = base + (size_t)col * ld + kb + hi * 8;
  v8bf lo = *(const v8bf*)(p);
  v8bf hh = *(const v8bf*)(p + 16);
  return __builtin_shufflevector(lo, hh, 0,1,2,3,4,5,6,7,8,9,10,11,12,13,14,15);
}

// ---------------- prep kernels ----------------
__global__ void f2bf_kernel(const float* __restrict__ s, __bf16* __restrict__ d, int n) {
  int i = blockIdx.x * 256 + threadIdx.x;
  if (i < n) d[i] = (__bf16)s[i];
}

// dst[n*K + k] = src[k*N + n]   (transpose + convert to bf16, N-major output)
__global__ void trans2bf_kernel(const float* __restrict__ s, __bf16* __restrict__ d, int K, int N) {
  int t = blockIdx.x * 256 + threadIdx.x;
  if (t >= K * N) return;
  int n = t % N, k = t / N;
  d[(size_t)n * K + k] = (__bf16)s[(size_t)k * N + n];
}

// u = x @ lora_A  (8192 x 4), one wave per row, fp32
__global__ __launch_bounds__(256) void lora_u_kernel(const float* __restrict__ x,
                                                     const float* __restrict__ Aq,
                                                     const float* __restrict__ Av,
                                                     float* __restrict__ uq, float* __restrict__ uv) {
  int lane = threadIdx.x & 31, wv = threadIdx.x >> 5;
  int row = blockIdx.x * 8 + wv;
  float aq[4] = {0.f,0.f,0.f,0.f}, av[4] = {0.f,0.f,0.f,0.f};
  for (int c = lane; c < CC; c += 32) {
    float xv = x[(size_t)row * CC + c];
#pragma unroll
    for (int t = 0; t < 4; ++t) { aq[t] += xv * Aq[c*4+t]; av[t] += xv * Av[c*4+t]; }
  }
#pragma unroll
  for (int t = 0; t < 4; ++t)
    for (int off = 16; off; off >>= 1) { aq[t] += __shfl_xor(aq[t], off, 32); av[t] += __shfl_xor(av[t], off, 32); }
  if (lane == 0) {
#pragma unroll
    for (int t = 0; t < 4; ++t) { uq[row*4+t] = aq[t]; uv[row*4+t] = av[t]; }
  }
}

// ---------------- shared GEMM helpers ----------------
static __device__ __forceinline__ void gemm_loads(const __bf16* Ab, const __bf16* Bb,
                                                  int m0, int n0, int kb, int lane,
                                                  v16bf a[2], v16bf b[4]) {
  int col = lane & 15;
  a[0] = load_fragA(Ab, CC, m0 + col, kb, lane);
  a[1] = load_fragA(Ab, CC, m0 + 16 + col, kb, lane);
#pragma unroll
  for (int nt = 0; nt < 4; ++nt) b[nt] = load_fragB(Bb, CC, n0 + nt*16 + col, kb, lane);
}

static __device__ __forceinline__ void gemm_phase(v8f acc[2][4], const v16bf a[2], const v16bf b[4]) {
#pragma unroll
  for (int nt = 0; nt < 4; ++nt) {
    acc[0][nt] = WMMA_BF16(a[0], b[nt], acc[0][nt]);
    acc[1][nt] = WMMA_BF16(a[1], b[nt], acc[1][nt]);
  }
}

// Two-phase (ping-pong register sets, no rotation copies) K mainloop.
static __device__ __forceinline__ void gemm_mainloop(const __bf16* Ab, const __bf16* Bb,
                                                     int m0, int n0, int lane, v8f acc[2][4]) {
  v16bf a0[2], b0[4], a1[2], b1[4];
  gemm_loads(Ab, Bb, m0, n0, 0, lane, a0, b0);
#pragma unroll
  for (int kb = 0; kb < CC; kb += 64) {
    gemm_loads(Ab, Bb, m0, n0, kb + 32, lane, a1, b1);
    gemm_phase(acc, a0, b0);
    if (kb + 64 < CC) gemm_loads(Ab, Bb, m0, n0, kb + 64, lane, a0, b0);
    gemm_phase(acc, a1, b1);
  }
}

// ---------------- fused QKV GEMM (bf16 WMMA) + bias + LoRA epilogue ----------------
__global__ __launch_bounds__(128)
void qkv_gemm_kernel(const __bf16* __restrict__ xbf, const __bf16* __restrict__ wT,
                     const float* __restrict__ bias,
                     const float* __restrict__ uq, const float* __restrict__ uv,
                     const float* __restrict__ Bq, const float* __restrict__ Bv,
                     __bf16* __restrict__ qbuf, __bf16* __restrict__ kbuf, __bf16* __restrict__ vT) {
  int lane = threadIdx.x & 31;
  int wv = threadIdx.x >> 5;
  int m0 = blockIdx.x * 128 + wv * 32;
  int n0 = blockIdx.y * 64;
  int col = lane & 15;
  v8f acc[2][4] = {};
  gemm_mainloop(xbf, wT, m0, n0, lane, acc);

  int hi = lane >> 4;
#pragma unroll
  for (int sub = 0; sub < 2; ++sub) {
#pragma unroll
    for (int nt = 0; nt < 4; ++nt) {
      int n = n0 + nt*16 + col;
      int which = n / CC;
      int rem = n - which * CC;
      int head = rem >> 6, d = rem & 63;
      float bval = bias[n];
#pragma unroll
      for (int g = 0; g < 8; ++g) {
        int r = m0 + sub*16 + g + hi*8;
        int bidx = r >> 10, s = r & 1023;
        int bh = bidx * NH + head;
        float val = acc[sub][nt][g] + bval;
        if (which == 0) {
          float lq = uq[r*4+0]*Bq[rem]      + uq[r*4+1]*Bq[768+rem]
                   + uq[r*4+2]*Bq[1536+rem] + uq[r*4+3]*Bq[2304+rem];
          val += LORA_SCALE * lq;
          qbuf[((size_t)bh*HWD + s)*HD + d] = (__bf16)val;
        } else if (which == 1) {
          kbuf[((size_t)bh*HWD + s)*HD + d] = (__bf16)val;
        } else {
          float lv = uv[r*4+0]*Bv[rem]      + uv[r*4+1]*Bv[768+rem]
                   + uv[r*4+2]*Bv[1536+rem] + uv[r*4+3]*Bv[2304+rem];
          val += LORA_SCALE * lv;
          vT[((size_t)bh*HD + d)*HWD + s] = (__bf16)val;
        }
      }
    }
  }
}

// ---------------- decomposed rel-pos bias precompute ----------------
__global__ void relbias_kernel(const __bf16* __restrict__ qbuf,
                               const float* __restrict__ rph, const float* __restrict__ rpw,
                               float* __restrict__ relh, float* __restrict__ relw) {
  int t = blockIdx.x * 256 + threadIdx.x;           // 2 * 96*1024*32 total
  int half = (t >= BHD*HWD*32) ? 1 : 0;
  int r = t - half * (BHD*HWD*32);
  int bh = r >> 15;
  int r2 = r & 32767;
  int qi = r2 >> 5, kk = r2 & 31;
  int qh = qi >> 5, qw = qi & 31;
  int idx = (half ? qw : qh) - kk + 31;
  const float* tab = (half ? rpw : rph) + (size_t)idx * HD;
  const __bf16* qr = qbuf + ((size_t)bh*HWD + qi) * HD;
  float s = 0.f;
#pragma unroll 8
  for (int d = 0; d < HD; ++d) s += (float)qr[d] * tab[d];
  (half ? relw : relh)[((size_t)bh*HWD + qi)*32 + kk] = s;
}

// ---------------- flash attention: T = K.Q^T transposed-score trick ----------------
static __device__ __forceinline__ void load_kfrags(v16bf k[4], const __bf16* kb, int j, int i, int lane) {
  k[0] = load_fragA(kb, HD, j + i, 0, lane);
  k[1] = load_fragA(kb, HD, j + i, 32, lane);
  k[2] = load_fragA(kb, HD, j + 16 + i, 0, lane);
  k[3] = load_fragA(kb, HD, j + 16 + i, 32, lane);
}

static __device__ __forceinline__ void attn_chunk(int jc, int lane, int i, int hi,
                                                  const v16bf ka[4], v16bf bq0, v16bf bq1,
                                                  const __bf16* vb, const float* rh, const float* rw,
                                                  float& run_m, float& run_l, v8f o[4]) {
  v8f t0 = {}, t1 = {};
  t0 = WMMA_BF16(ka[0], bq0, t0);
  t0 = WMMA_BF16(ka[1], bq1, t0);
  t1 = WMMA_BF16(ka[2], bq0, t1);
  t1 = WMMA_BF16(ka[3], bq1, t1);

  // V-operand loads overlap the softmax VALU below
  v16bf bv[4];
#pragma unroll
  for (int dt = 0; dt < 4; ++dt) bv[dt] = load_fragB(vb, HWD, dt*16 + i, jc, lane);

  float rhv = rh[jc >> 5];   // kh constant over a 32-aligned key chunk
  float p0[8], p1[8];
  float tm = -1e30f;
#pragma unroll
  for (int g = 0; g < 8; ++g) {
    int j0 = hi*8 + g;       // kw = local j since jc % 32 == 0
    p0[g] = t0[g]*ATT_SCALE + rhv + rw[j0];
    p1[g] = t1[g]*ATT_SCALE + rhv + rw[16 + j0];
    tm = fmaxf(tm, fmaxf(p0[g], p1[g]));
  }
  tm = fmaxf(tm, __shfl_xor(tm, 16, 32));       // lanes L and L^16 share query i
  float mnew = fmaxf(run_m, tm);
  float corr = __expf(run_m - mnew);
  float ps = 0.f;
#pragma unroll
  for (int g = 0; g < 8; ++g) {
    p0[g] = __expf(p0[g] - mnew);
    p1[g] = __expf(p1[g] - mnew);
    ps += p0[g] + p1[g];
  }
  ps += __shfl_xor(ps, 16, 32);
  run_l = run_l * corr + ps;
  run_m = mnew;

  // P lands directly in A-operand layout (transposed-score trick)
  v16bf pa;
#pragma unroll
  for (int g = 0; g < 8; ++g) { pa[g] = (__bf16)p0[g]; pa[8+g] = (__bf16)p1[g]; }

  float cq[8];
#pragma unroll
  for (int g = 0; g < 8; ++g) cq[g] = __shfl(corr, g + hi*8, 32);  // per-output-row corr
#pragma unroll
  for (int dt = 0; dt < 4; ++dt) {
#pragma unroll
    for (int g = 0; g < 8; ++g) o[dt][g] *= cq[g];
    o[dt] = WMMA_BF16(pa, bv[dt], o[dt]);
  }
}

__global__ __launch_bounds__(128)
void attn_kernel(const __bf16* __restrict__ qbuf, const __bf16* __restrict__ kbuf,
                 const __bf16* __restrict__ vT, const float* __restrict__ relh,
                 const float* __restrict__ relw, __bf16* __restrict__ outc) {
  int lane = threadIdx.x & 31;
  int wv = threadIdx.x >> 5;
  int bh = blockIdx.y;
  int q0 = blockIdx.x * 64 + wv * 16;
  int i = lane & 15, hi = lane >> 4;

  const __bf16* qb = qbuf + (size_t)bh * HWD * HD;
  const __bf16* kb = kbuf + (size_t)bh * HWD * HD;
  const __bf16* vb = vT   + (size_t)bh * HD * HWD;
  int gq = q0 + i;
  v16bf bq0 = load_fragB(qb, HD, gq, 0, lane);    // Q^T B-operand, K = head-dim
  v16bf bq1 = load_fragB(qb, HD, gq, 32, lane);
  const float* rh = relh + ((size_t)bh*HWD + gq) * 32;
  const float* rw = relw + ((size_t)bh*HWD + gq) * 32;

  float run_m = -1e30f, run_l = 0.f;
  v8f o[4] = {};

  // Two-phase ping-pong over 32-key chunks (no register rotation).
  v16bf ka0[4], ka1[4];
  load_kfrags(ka0, kb, 0, i, lane);
#pragma unroll 4
  for (int jc = 0; jc < HWD; jc += 64) {
    load_kfrags(ka1, kb, jc + 32, i, lane);
    attn_chunk(jc, lane, i, hi, ka0, bq0, bq1, vb, rh, rw, run_m, run_l, o);
    if (jc + 64 < HWD) load_kfrags(ka0, kb, jc + 64, i, lane);
    attn_chunk(jc + 32, lane, i, hi, ka1, bq0, bq1, vb, rh, rw, run_m, run_l, o);
  }

  float il = 1.f / run_l;
  float cq[8];
#pragma unroll
  for (int g = 0; g < 8; ++g) cq[g] = __shfl(il, g + hi*8, 32);
  int b = bh / NH, head = bh - b * NH;
#pragma unroll
  for (int dt = 0; dt < 4; ++dt) {
#pragma unroll
    for (int g = 0; g < 8; ++g) {
      int s = q0 + g + hi*8;
      int c = head*HD + dt*16 + i;
      outc[((size_t)(b*HWD + s))*CC + c] = (__bf16)(o[dt][g] * cq[g]);
    }
  }
}

// ---------------- output projection GEMM ----------------
__global__ __launch_bounds__(128)
void proj_gemm_kernel(const __bf16* __restrict__ A, const __bf16* __restrict__ wT,
                      const float* __restrict__ bias, float* __restrict__ out) {
  int lane = threadIdx.x & 31;
  int wv = threadIdx.x >> 5;
  int m0 = blockIdx.x * 128 + wv * 32;
  int n0 = blockIdx.y * 64;
  int col = lane & 15;
  v8f acc[2][4] = {};
  gemm_mainloop(A, wT, m0, n0, lane, acc);

  int hi = lane >> 4;
#pragma unroll
  for (int sub = 0; sub < 2; ++sub) {
#pragma unroll
    for (int nt = 0; nt < 4; ++nt) {
      int n = n0 + nt*16 + col;
      float bval = bias[n];
#pragma unroll
      for (int g = 0; g < 8; ++g) {
        int r = m0 + sub*16 + g + hi*8;
        out[(size_t)r * CC + n] = acc[sub][nt][g] + bval;
      }
    }
  }
}

extern "C" void kernel_launch(void* const* d_in, const int* in_sizes, int n_in,
                              void* d_out, int out_size, void* d_ws, size_t ws_size,
                              hipStream_t stream) {
  (void)in_sizes; (void)n_in; (void)out_size; (void)ws_size;
  const float* x      = (const float*)d_in[0];
  const float* qkv_w  = (const float*)d_in[1];
  const float* qkv_b  = (const float*)d_in[2];
  const float* proj_w = (const float*)d_in[3];
  const float* proj_b = (const float*)d_in[4];
  const float* Aq     = (const float*)d_in[5];
  const float* Bq     = (const float*)d_in[6];
  const float* Av     = (const float*)d_in[7];
  const float* Bv     = (const float*)d_in[8];
  const float* rph    = (const float*)d_in[9];
  const float* rpw    = (const float*)d_in[10];
  float* out = (float*)d_out;

  char* ws = (char*)d_ws;
  size_t off = 0;
  auto carve = [&](size_t bytes) { char* p = ws + off; off += (bytes + 255) & ~(size_t)255; return p; };

  __bf16* xbf   = (__bf16*)carve((size_t)MROWS * CC * 2);
  __bf16* qkvT  = (__bf16*)carve((size_t)NQKV * CC * 2);
  __bf16* projT = (__bf16*)carve((size_t)CC * CC * 2);
  float*  uq    = (float*) carve((size_t)MROWS * 4 * 4);
  float*  uv    = (float*) carve((size_t)MROWS * 4 * 4);
  __bf16* qbuf  = (__bf16*)carve((size_t)BHD * HWD * HD * 2);
  __bf16* kbuf  = (__bf16*)carve((size_t)BHD * HWD * HD * 2);
  __bf16* vTb   = (__bf16*)carve((size_t)BHD * HD * HWD * 2);
  float*  relh  = (float*) carve((size_t)BHD * HWD * 32 * 4);
  float*  relw  = (float*) carve((size_t)BHD * HWD * 32 * 4);
  __bf16* outc  = (__bf16*)carve((size_t)MROWS * CC * 2);

  int nx = MROWS * CC;
  f2bf_kernel<<<(nx + 255) / 256, 256, 0, stream>>>(x, xbf, nx);
  trans2bf_kernel<<<(CC * NQKV + 255) / 256, 256, 0, stream>>>(qkv_w, qkvT, CC, NQKV);
  trans2bf_kernel<<<(CC * CC + 255) / 256, 256, 0, stream>>>(proj_w, projT, CC, CC);
  lora_u_kernel<<<MROWS / 8, 256, 0, stream>>>(x, Aq, Av, uq, uv);
  qkv_gemm_kernel<<<dim3(MROWS / 128, NQKV / 64), 128, 0, stream>>>(
      xbf, qkvT, qkv_b, uq, uv, Bq, Bv, qbuf, kbuf, vTb);
  relbias_kernel<<<(2 * BHD * HWD * 32) / 256, 256, 0, stream>>>(qbuf, rph, rpw, relh, relw);
  attn_kernel<<<dim3(HWD / 64, BHD), 128, 0, stream>>>(qbuf, kbuf, vTb, relh, relw, outc);
  proj_gemm_kernel<<<dim3(MROWS / 128, CC / 64), 128, 0, stream>>>(outc, projT, proj_b, out);
}
